// Generator_WGAN_GP_Upsample_SmallLinear_35321811043091
// MI455X (gfx1250) — matrix operand order repaired and checked by reading, not compiled, not run on hardware
//
#include <hip/hip_runtime.h>
#include <hip/hip_bf16.h>

// ---------------------------------------------------------------------------
// Types for WMMA (CDNA5 / gfx1250, wave32)
// ---------------------------------------------------------------------------
typedef __attribute__((ext_vector_type(16))) __bf16 v16bf;
typedef __attribute__((ext_vector_type(8)))  float  v8f;

union ABPack { uint4 q[2]; unsigned int u[8]; v16bf v; };

// ---------------------------------------------------------------------------
// Helpers
// ---------------------------------------------------------------------------
__device__ __forceinline__ unsigned short f2bf(float f) {
    unsigned int u = __float_as_uint(f);
    u = (u + 0x7FFFu + ((u >> 16) & 1u)) >> 16;           // round-to-nearest-even
    return (unsigned short)u;
}
__device__ __forceinline__ float bf2f(unsigned short h) {
    return __uint_as_float(((unsigned int)h) << 16);
}
__device__ __forceinline__ float lrelu(float v) { return v >= 0.f ? v : 0.2f * v; }

// bicubic weights, A = -0.75 (PyTorch)
__device__ __forceinline__ float bc_c1(float u) { return (1.25f * u - 2.25f) * u * u + 1.f; }
__device__ __forceinline__ float bc_c2(float u) { return ((-0.75f * u + 3.75f) * u - 6.f) * u + 3.f; }

// ---------------------------------------------------------------------------
// f32 -> bf16 pack
// ---------------------------------------------------------------------------
__global__ void pack_bf16_kernel(const float* __restrict__ in,
                                 unsigned short* __restrict__ out, int n) {
    int i = blockIdx.x * 256 + threadIdx.x;
    if (i < n) out[i] = f2bf(in[i]);
}

// ---------------------------------------------------------------------------
// GEMM: oi[b,n] = sum_k x[b,k] * Wi[n,k] + bi[n]
// M=4096, K=160, N=936 (59 n-tiles, last partially masked on store only:
// out-of-range columns load clamped-row garbage that stays in their own
// D column and is never stored / never counted).
// One wave per 16x16 tile, 8 waves/block, 1888 blocks.
// A/B per-lane data is contiguous in K -> two b128 loads per tile chunk.
// Also emits per-block (sum, sumsq) partials for BN0 batch stats.
// ---------------------------------------------------------------------------
__global__ void gemm_wmma_kernel(const unsigned short* __restrict__ xbf,
                                 const unsigned short* __restrict__ wibf,
                                 const float* __restrict__ bias,
                                 float* __restrict__ out,
                                 float* __restrict__ partials) {
    __shared__ float red[16];
    const int t    = threadIdx.x;
    const int wv   = t >> 5;
    const int lane = t & 31;
    const int hi   = lane >> 4;        // K-half selector
    const int ln   = lane & 15;

    const int tile = blockIdx.x * 8 + wv;      // 0..15103
    const int mt   = tile / 59;
    const int nt   = tile % 59;

    const int m    = mt * 16 + ln;             // A row (batch), always valid
    const int n    = nt * 16 + ln;             // B row = output feature
    const bool nok = (n < 936);
    const int  nc  = nok ? n : 935;            // clamped (address validity only)

    v8f acc;
    {
        const float bv = bias[nc];
        #pragma unroll
        for (int r = 0; r < 8; ++r) acc[r] = bv;
    }

    // row base pointers, 16B aligned (rows are 320B)
    const uint4* xa4 = (const uint4*)(xbf  + (size_t)m  * 160);
    const uint4* wa4 = (const uint4*)(wibf + (size_t)nc * 160);

    #pragma unroll
    for (int kc = 0; kc < 160; kc += 32) {
        ABPack A, B;
        // A 16x32 bf16: elems 0..7 -> K = kc + 8*hi + {0..7}
        //               elems 8..15 -> K = kc + 16 + 8*hi + {0..7}
        A.q[0] = xa4[(kc + 8 * hi) >> 3];
        A.q[1] = xa4[(kc + 16 + 8 * hi) >> 3];
        // B 32x16 bf16: elems 0..7 -> K = kc + 8*hi + {0..7}
        //               elems 8..15 -> K = kc + 16 + 8*hi + {0..7}
        B.q[0] = wa4[(kc + 8 * hi) >> 3];
        B.q[1] = wa4[(kc + 16 + 8 * hi) >> 3];

        acc = __builtin_amdgcn_wmma_f32_16x16x32_bf16(
            /*neg_a=*/false, A.v, /*neg_b=*/false, B.v,
            /*c_mod=*/(short)0, acc, /*reuse_a=*/false, /*reuse_b=*/false);
    }

    float s = 0.f, sq = 0.f;
    #pragma unroll
    for (int r = 0; r < 8; ++r) {
        const int mm = mt * 16 + r + 8 * hi;   // D: VGPR r -> M = r + 8*hi
        if (nok) {
            const float v = acc[r];
            out[(size_t)mm * 936 + n] = v;
            s  += v;
            sq += v * v;
        }
    }
    // wave reduce
    #pragma unroll
    for (int off = 16; off > 0; off >>= 1) {
        s  += __shfl_down(s,  off, 32);
        sq += __shfl_down(sq, off, 32);
    }
    if (lane == 0) { red[wv * 2] = s; red[wv * 2 + 1] = sq; }
    __syncthreads();
    if (t == 0) {
        float a = 0.f, b = 0.f;
        #pragma unroll
        for (int w = 0; w < 8; ++w) { a += red[w * 2]; b += red[w * 2 + 1]; }
        partials[(size_t)blockIdx.x * 2 + 0] = a;
        partials[(size_t)blockIdx.x * 2 + 1] = b;
    }
}

// ---------------------------------------------------------------------------
// Reduce per-block partials -> stats[0..nch-1]=mean, stats[nch..2nch-1]=invstd
// Partials layout: [nblocks][2*nch] (sums first, then sumsqs).
// Requires 2*nch divides 256.
// ---------------------------------------------------------------------------
__global__ void finalize_stats_kernel(const float* __restrict__ partials,
                                      int nblocks, float invN, int nch,
                                      float* __restrict__ stats) {
    __shared__ float red[256];
    const int t  = threadIdx.x;
    const int tw = 2 * nch;
    const int i  = t % tw;
    const int k  = t / tw;
    const int stride = 256 / tw;
    float a = 0.f;
    for (int blk = k; blk < nblocks; blk += stride)
        a += partials[(size_t)blk * tw + i];
    red[t] = a;
    __syncthreads();
    if (t < tw) {
        float sft = 0.f;
        for (int kk = 0; kk < stride; ++kk) sft += red[kk * tw + t];
        red[t] = sft;
    }
    __syncthreads();
    if (t < nch) {
        const float mean = red[t] * invN;
        const float var  = red[nch + t] * invN - mean * mean;
        stats[t]       = mean;
        stats[nch + t] = rsqrtf(var + 1e-5f);
    }
}

// ---------------------------------------------------------------------------
// Build h0 [B,9,117] (bf16): channels 0..7 = lrelu(BN0(oi)), channel 8 = class
// branch (emb[y] @ Wc^T + bc). One block per batch sample, 128 threads.
// ---------------------------------------------------------------------------
__global__ void build_h0_kernel(const float* __restrict__ oi,
                                const float* __restrict__ stats0,
                                const float* __restrict__ g0,
                                const float* __restrict__ b0,
                                const int* __restrict__ y,
                                const float* __restrict__ emb,
                                const float* __restrict__ Wc,
                                const float* __restrict__ bc,
                                unsigned short* __restrict__ h0) {
    __shared__ float se[50];
    const int b = blockIdx.x;
    const int j = threadIdx.x;
    if (j < 50) se[j] = emb[y[b] * 50 + j];
    __syncthreads();
    if (j >= 117) return;

    const float m  = stats0[0];
    const float is = stats0[1];
    const float gs  = g0[0] * is;
    const float ofs = b0[0] - m * gs;

    #pragma unroll
    for (int c = 0; c < 8; ++c) {
        float v = oi[(size_t)b * 936 + c * 117 + j];
        v = lrelu(v * gs + ofs);
        h0[((size_t)b * 9 + c) * 117 + j] = f2bf(v);
    }
    float acc = bc[j];
    #pragma unroll 10
    for (int k = 0; k < 50; ++k) acc += se[k] * Wc[j * 50 + k];
    h0[((size_t)b * 9 + 8) * 117 + j] = f2bf(acc);
}

// ---------------------------------------------------------------------------
// Stage kernel: (optional BN+lrelu on load) -> bicubic up -> 1x1 conv ->
// bf16 store of pre-activation + per-block channel stats partials [blk][16].
// One thread per (b, l_out); B*L_out is always a multiple of 256 here.
// ---------------------------------------------------------------------------
template <int CIN, bool HASBN>
__global__ void stage_kernel(const unsigned short* __restrict__ in_bf, int L_in,
                             const float* __restrict__ stats,
                             const float* __restrict__ gch,
                             const float* __restrict__ bch,
                             const float* __restrict__ wconv,
                             unsigned short* __restrict__ out_bf, int L_out,
                             float* __restrict__ partials) {
    __shared__ float sW[8 * CIN];
    __shared__ float sScale[CIN];
    __shared__ float sOff[CIN];
    __shared__ float red[128];

    const int t = threadIdx.x;
    if (t < 8 * CIN) sW[t] = wconv[t];
    if (HASBN) {
        if (t < CIN) {                       // CIN == 8 whenever HASBN
            const float is = stats[8 + t];
            const float gs = gch[t] * is;
            sScale[t] = gs;
            sOff[t]   = bch[t] - stats[t] * gs;
        }
    }
    __syncthreads();

    const int idx = blockIdx.x * 256 + t;
    const int l = idx % L_out;
    const int b = idx / L_out;

    const float scale = (float)L_in / (float)L_out;
    const float src = ((float)l + 0.5f) * scale - 0.5f;
    const float fi  = floorf(src);
    const float tt  = src - fi;
    const int   i0  = (int)fi;
    float wt[4];
    wt[0] = bc_c2(tt + 1.f);
    wt[1] = bc_c1(tt);
    wt[2] = bc_c1(1.f - tt);
    wt[3] = bc_c2(2.f - tt);

    float interp[CIN];
    #pragma unroll
    for (int c = 0; c < CIN; ++c) interp[c] = 0.f;

    const size_t base = (size_t)b * CIN * L_in;
    #pragma unroll
    for (int j = 0; j < 4; ++j) {
        int ij = i0 - 1 + j;
        ij = ij < 0 ? 0 : (ij >= L_in ? L_in - 1 : ij);
        #pragma unroll
        for (int c = 0; c < CIN; ++c) {
            float v = bf2f(in_bf[base + (size_t)c * L_in + ij]);
            if (HASBN) v = lrelu(v * sScale[c] + sOff[c]);
            interp[c] += wt[j] * v;
        }
    }

    float sums[8], sqs[8];
    const size_t ob = (size_t)b * 8 * L_out + l;
    #pragma unroll
    for (int o = 0; o < 8; ++o) {
        float a = 0.f;
        #pragma unroll
        for (int c = 0; c < CIN; ++c) a += sW[o * CIN + c] * interp[c];
        out_bf[ob + (size_t)o * L_out] = f2bf(a);
        sums[o] = a;
        sqs[o]  = a * a;
    }

    // wave32 reduce of 8 sums + 8 sumsqs
    #pragma unroll
    for (int off = 16; off > 0; off >>= 1) {
        #pragma unroll
        for (int o = 0; o < 8; ++o) {
            sums[o] += __shfl_down(sums[o], off, 32);
            sqs[o]  += __shfl_down(sqs[o],  off, 32);
        }
    }
    const int lane = t & 31, wv = t >> 5;
    if (lane == 0) {
        #pragma unroll
        for (int o = 0; o < 8; ++o) {
            red[wv * 16 + o]     = sums[o];
            red[wv * 16 + 8 + o] = sqs[o];
        }
    }
    __syncthreads();
    if (t < 16) {
        float a = 0.f;
        #pragma unroll
        for (int w = 0; w < 8; ++w) a += red[w * 16 + t];
        partials[(size_t)blockIdx.x * 16 + t] = a;
    }
}

// ---------------------------------------------------------------------------
// Final: out[b,l] = sum_o w6[o] * lrelu(BN5(in[b,o,l]))   (f32 output)
// ---------------------------------------------------------------------------
__global__ void final_kernel(const unsigned short* __restrict__ in_bf,
                             const float* __restrict__ stats,
                             const float* __restrict__ g,
                             const float* __restrict__ bb,
                             const float* __restrict__ w6,
                             float* __restrict__ out) {
    const int idx = blockIdx.x * 256 + threadIdx.x;
    const int l = idx % 3750;
    const int b = idx / 3750;
    const size_t base = (size_t)b * 8 * 3750 + l;
    float acc = 0.f;
    #pragma unroll
    for (int o = 0; o < 8; ++o) {
        const float is = stats[8 + o];
        const float gs = g[o] * is;
        const float of = bb[o] - stats[o] * gs;
        float v = bf2f(in_bf[base + (size_t)o * 3750]) * gs + of;
        acc += w6[o] * lrelu(v);
    }
    out[(size_t)b * 3750 + l] = acc;
}

// ---------------------------------------------------------------------------
// Host launch
// ---------------------------------------------------------------------------
extern "C" void kernel_launch(void* const* d_in, const int* in_sizes, int n_in,
                              void* d_out, int out_size, void* d_ws, size_t ws_size,
                              hipStream_t stream) {
    (void)in_sizes; (void)n_in; (void)out_size; (void)ws_size;

    const float* x   = (const float*)d_in[0];
    const int*   y   = (const int*)  d_in[1];
    const float* emb = (const float*)d_in[2];
    const float* Wc  = (const float*)d_in[3];
    const float* bc  = (const float*)d_in[4];
    const float* Wi  = (const float*)d_in[5];
    const float* bi  = (const float*)d_in[6];
    const float* g0  = (const float*)d_in[7];
    const float* b0  = (const float*)d_in[8];
    const float* w1  = (const float*)d_in[9];
    const float* w2  = (const float*)d_in[10];
    const float* w3  = (const float*)d_in[11];
    const float* w4  = (const float*)d_in[12];
    const float* w5  = (const float*)d_in[13];
    const float* w6  = (const float*)d_in[14];
    const float* g1  = (const float*)d_in[15];
    const float* b1  = (const float*)d_in[16];
    const float* g2  = (const float*)d_in[17];
    const float* b2  = (const float*)d_in[18];
    const float* g3  = (const float*)d_in[19];
    const float* b3  = (const float*)d_in[20];
    const float* g4  = (const float*)d_in[21];
    const float* b4  = (const float*)d_in[22];
    const float* g5  = (const float*)d_in[23];
    const float* b5  = (const float*)d_in[24];
    float* out = (float*)d_out;

    const int B = 4096;

    // workspace layout (256B aligned slabs)
    char* ws = (char*)d_ws;
    size_t off = 0;
    auto alloc = [&](size_t bytes) -> char* {
        char* p = ws + off;
        off += (bytes + 255) & ~(size_t)255;
        return p;
    };
    unsigned short* xbf   = (unsigned short*)alloc((size_t)B * 160 * 2);
    unsigned short* wibf  = (unsigned short*)alloc((size_t)936 * 160 * 2);
    float*          oi    = (float*)         alloc((size_t)B * 936 * 4);
    unsigned short* h0    = (unsigned short*)alloc((size_t)B * 9 * 117 * 2);
    float*          stats = (float*)         alloc((size_t)6 * 16 * 4);
    float*          parts = (float*)         alloc((size_t)60000 * 16 * 4);
    unsigned short* bufA  = (unsigned short*)alloc((size_t)B * 8 * 3750 * 2);
    unsigned short* bufB  = (unsigned short*)alloc((size_t)B * 8 * 1875 * 2);

    float* stats0 = stats;
    float* stats1 = stats + 16;
    float* stats2 = stats + 32;
    float* stats3 = stats + 48;
    float* stats4 = stats + 64;
    float* stats5 = stats + 80;

    // 1) pack f32 -> bf16
    {
        const int nx = B * 160;
        pack_bf16_kernel<<<(nx + 255) / 256, 256, 0, stream>>>(x, xbf, nx);
        const int nw = 936 * 160;
        pack_bf16_kernel<<<(nw + 255) / 256, 256, 0, stream>>>(Wi, wibf, nw);
    }

    // 2) WMMA GEMM + BN0 partials
    gemm_wmma_kernel<<<1888, 256, 0, stream>>>(xbf, wibf, bi, oi, parts);
    finalize_stats_kernel<<<1, 256, 0, stream>>>(
        parts, 1888, 1.f / ((float)B * 936.f), 1, stats0);

    // 3) h0 = concat(lrelu(BN0(oi)).reshape(B,8,117), class branch)
    build_h0_kernel<<<B, 128, 0, stream>>>(oi, stats0, g0, b0, y, emb, Wc, bc, h0);

    // 4) stages
    // stage 1: h0 (activated, CIN=9) -> bufA pre-act, L 117 -> 234, conv w1
    {
        const int nb = B * 234 / 256;   // 3744
        stage_kernel<9, false><<<nb, 256, 0, stream>>>(
            h0, 117, stats0, g0, b0, w1, bufA, 234, parts);
        finalize_stats_kernel<<<1, 256, 0, stream>>>(
            parts, nb, 1.f / ((float)B * 234.f), 8, stats1);
    }
    // stage 2: bufA (BN1) -> bufB, 234 -> 468, conv w2
    {
        const int nb = B * 468 / 256;   // 7488
        stage_kernel<8, true><<<nb, 256, 0, stream>>>(
            bufA, 234, stats1, g1, b1, w2, bufB, 468, parts);
        finalize_stats_kernel<<<1, 256, 0, stream>>>(
            parts, nb, 1.f / ((float)B * 468.f), 8, stats2);
    }
    // stage 3: bufB (BN2) -> bufA, 468 -> 937, conv w3
    {
        const int nb = B * 937 / 256;   // 14992
        stage_kernel<8, true><<<nb, 256, 0, stream>>>(
            bufB, 468, stats2, g2, b2, w3, bufA, 937, parts);
        finalize_stats_kernel<<<1, 256, 0, stream>>>(
            parts, nb, 1.f / ((float)B * 937.f), 8, stats3);
    }
    // stage 4: bufA (BN3) -> bufB, 937 -> 1875, conv w4
    {
        const int nb = B * 1875 / 256;  // 30000
        stage_kernel<8, true><<<nb, 256, 0, stream>>>(
            bufA, 937, stats3, g3, b3, w4, bufB, 1875, parts);
        finalize_stats_kernel<<<1, 256, 0, stream>>>(
            parts, nb, 1.f / ((float)B * 1875.f), 8, stats4);
    }
    // stage 5: bufB (BN4) -> bufA, 1875 -> 3750, conv w5
    {
        const int nb = B * 3750 / 256;  // 60000
        stage_kernel<8, true><<<nb, 256, 0, stream>>>(
            bufB, 1875, stats4, g4, b4, w5, bufA, 3750, parts);
        finalize_stats_kernel<<<1, 256, 0, stream>>>(
            parts, nb, 1.f / ((float)B * 3750.f), 8, stats5);
    }

    // 5) final conv 8->1 with BN5+lrelu on load, f32 output
    final_kernel<<<B * 3750 / 256, 256, 0, stream>>>(bufA, stats5, g5, b5, w6, out);
}
